// AttnBlock_18837726560987
// MI455X (gfx1250) — matrix operand order repaired and checked by reading, not compiled, not run on hardware
//
#include <hip/hip_runtime.h>

// Problem dims (fixed by the reference): B=16, C=512, H=W=32, HW=1024, 3C=1536
#define NB   16
#define CCH  512
#define HW   1024
#define OC3  1536
#define PW   34            // padded width/height (1-pixel zero halo)
#define PHW  (PW * PW)     // 1156 padded pixels

typedef __bf16 bf16_t;
typedef __attribute__((ext_vector_type(16))) __bf16 v16bf;
typedef __attribute__((ext_vector_type(8)))  __bf16 v8bf;
typedef __attribute__((ext_vector_type(8)))  float  v8f;

// ---------- helpers ----------

__device__ __forceinline__ v8f wmma_bf16(v16bf a, v16bf b, v8f c) {
    // D = A(16x32 bf16) * B(32x16 bf16) + C(16x16 f32)
    return __builtin_amdgcn_wmma_f32_16x16x32_bf16(
        /*neg_a=*/false, a, /*neg_b=*/false, b,
        /*c_mod=*/(short)0, c, /*reuse_a=*/false, /*reuse_b=*/false);
}

// Assemble an A fragment (16 rows x 32 K) from two 16-byte chunks.
// ISA layout: lanes 0-15 = rows, K {0..7,16..23}; lanes 16-31 = rows, K {8..15,24..31}.
__device__ __forceinline__ v16bf make_a(v8bf lo, v8bf hi) {
    v16bf r;
#pragma unroll
    for (int i = 0; i < 8; ++i) { r[i] = lo[i]; r[i + 8] = hi[i]; }
    return r;
}

// A fragment from global memory, row stride ld.
__device__ __forceinline__ v16bf load_a_frag_g(const bf16_t* base, int ld, int lane) {
    int row = lane & 15, half = lane >> 4;
    const bf16_t* p = base + (size_t)row * ld + half * 8;
    return make_a(*(const v8bf*)p, *(const v8bf*)(p + 16));
}

// B fragment: 32 K x 16 cols, B[k][col] = mem[col*ld + k].
__device__ __forceinline__ v16bf load_b_frag(const bf16_t* base, int ld, int lane) {
    int col = lane & 15, half = lane >> 4;
    const bf16_t* p = base + (size_t)col * ld + half * 8;
    return make_a(*(const v8bf*)p, *(const v8bf*)(p + 16));
}

__device__ __forceinline__ float half16_max(float v) {
#pragma unroll
    for (int off = 1; off < 16; off <<= 1) v = fmaxf(v, __shfl_xor(v, off, 16));
    return v;
}
__device__ __forceinline__ float half16_sum(float v) {
#pragma unroll
    for (int off = 1; off < 16; off <<= 1) v += __shfl_xor(v, off, 16);
    return v;
}

// ---------- pack / convert kernels ----------

// x [B][C][H][W] fp32 -> xb [B][34*34][C] bf16 with zero halo.
__global__ void pack_x_kernel(const float* __restrict__ x, bf16_t* __restrict__ xb) {
    size_t i = (size_t)blockIdx.x * blockDim.x + threadIdx.x;   // over NB*PHW*CCH
    if (i >= (size_t)NB * PHW * CCH) return;
    int c = (int)(i & (CCH - 1));
    size_t bp = i >> 9;
    int p = (int)(bp % PHW);
    int b = (int)(bp / PHW);
    int y = p / PW, xx = p % PW;
    float v = 0.f;
    if (y >= 1 && y <= 32 && xx >= 1 && xx <= 32)
        v = x[((size_t)b * CCH + c) * HW + (size_t)(y - 1) * 32 + (xx - 1)];
    xb[i] = (bf16_t)v;
}

__global__ void pack_wqkv_kernel(const float* __restrict__ w, bf16_t* __restrict__ wb) {
    size_t i = (size_t)blockIdx.x * blockDim.x + threadIdx.x;   // over 3C*C*9
    if (i >= (size_t)OC3 * CCH * 9) return;
    int t = (int)(i % 9);
    size_t r = i / 9;
    int c  = (int)(r & (CCH - 1));
    int oc = (int)(r >> 9);
    wb[((size_t)t * OC3 + oc) * CCH + c] = (bf16_t)w[i];   // wb[tap][oc][c]
}

__global__ void pack_wp_kernel(const float* __restrict__ w, bf16_t* __restrict__ wb) {
    size_t i = (size_t)blockIdx.x * blockDim.x + threadIdx.x;   // over C*C
    if (i >= (size_t)CCH * CCH) return;
    wb[i] = (bf16_t)w[i];
}

// ---------- GEMM1: qkv = conv3x3(x) as 9 shifted 1x1 GEMMs over the padded image ----
// 512 threads = 16 waves. Block: (b, oc-tile of 32). Wave w covers pixels
// [w*64, w*64+64) as 4 tiles; per tile the padded-image offset is a compile-time
// immediate. Each B fragment feeds 2 WMMAs (oc rows 0-15 / 16-31).
__global__ void qkv_conv_kernel(const bf16_t* __restrict__ xb,
                                const bf16_t* __restrict__ wqkvb,
                                const float*  __restrict__ b_qkv,
                                bf16_t* __restrict__ qkT,
                                bf16_t* __restrict__ vbuf) {
    int b    = blockIdx.x;
    int oc0  = blockIdx.y * 32;
    int wave = threadIdx.x >> 5;
    int lane = threadIdx.x & 31;
    int col  = lane & 15;
    int half = lane >> 4;

    const bf16_t* xbb = xb + (size_t)b * PHW * CCH;

    // tile t (t=0..3): pixel n = wave*64 + t*16 + col
    //   y = wave*2 + (t>>1), x = (t&1)*16 + col
    // padded offset = ((y+dy)*34 + x+dx)*CCH; tile-relative part is constant:
    const int TOFF0 = 0;
    const int TOFF1 = 16 * CCH;                 //  (t&1)=1
    const int TOFF2 = PW * CCH;                 //  (t>>1)=1
    const int TOFF3 = (PW + 16) * CCH;

    v8f acc0[4] = {};
    v8f acc1[4] = {};
#pragma unroll 1
    for (int tap = 0; tap < 9; ++tap) {
        int dy = tap / 3, dx = tap % 3;          // shift into the padded image
        const bf16_t* wb = wqkvb + ((size_t)tap * OC3 + oc0) * CCH;
        const bf16_t* base =
            xbb + (size_t)((wave * 2 + dy) * PW + col + dx) * CCH + half * 8;
#pragma unroll 1
        for (int k = 0; k < CCH; k += 32) {
            v16bf a0 = load_a_frag_g(wb + k, CCH, lane);
            v16bf a1 = load_a_frag_g(wb + (size_t)16 * CCH + k, CCH, lane);
            v16bf b0 = make_a(*(const v8bf*)(base + k + TOFF0), *(const v8bf*)(base + k + TOFF0 + 16));
            v16bf b1 = make_a(*(const v8bf*)(base + k + TOFF1), *(const v8bf*)(base + k + TOFF1 + 16));
            v16bf b2 = make_a(*(const v8bf*)(base + k + TOFF2), *(const v8bf*)(base + k + TOFF2 + 16));
            v16bf b3 = make_a(*(const v8bf*)(base + k + TOFF3), *(const v8bf*)(base + k + TOFF3 + 16));
            acc0[0] = wmma_bf16(a0, b0, acc0[0]);
            acc1[0] = wmma_bf16(a1, b0, acc1[0]);
            acc0[1] = wmma_bf16(a0, b1, acc0[1]);
            acc1[1] = wmma_bf16(a1, b1, acc1[1]);
            acc0[2] = wmma_bf16(a0, b2, acc0[2]);
            acc1[2] = wmma_bf16(a1, b2, acc1[2]);
            acc0[3] = wmma_bf16(a0, b3, acc0[3]);
            acc1[3] = wmma_bf16(a1, b3, acc1[3]);
        }
    }

    const float qscale = 0.044194173824159216f;  // 1/sqrt(512)
    bool is_q = (oc0 < CCH);
    bool is_v = (oc0 >= 2 * CCH);
    float bias0[8], bias1[8];
#pragma unroll
    for (int r = 0; r < 8; ++r) {
        bias0[r] = b_qkv[oc0 + r + 8 * half];
        bias1[r] = b_qkv[oc0 + 16 + r + 8 * half];
    }

    if (!is_v) {
#pragma unroll
        for (int t = 0; t < 4; ++t) {
            int n = wave * 64 + t * 16 + col;
            v8bf pk0, pk1;
#pragma unroll
            for (int r = 0; r < 8; ++r) {
                float v0 = acc0[t][r] + bias0[r];
                float v1 = acc1[t][r] + bias1[r];
                if (is_q) { v0 *= qscale; v1 *= qscale; }
                pk0[r] = (bf16_t)v0;
                pk1[r] = (bf16_t)v1;
            }
            bf16_t* row = qkT + ((size_t)(b * HW + n)) * 1024 + oc0 + 8 * half;
            *(v8bf*)(row)      = pk0;
            *(v8bf*)(row + 16) = pk1;
        }
    } else {
#pragma unroll
        for (int t = 0; t < 4; ++t) {
            int n = wave * 64 + t * 16 + col;
#pragma unroll
            for (int r = 0; r < 8; ++r) {
                int oc_a = oc0 + r + 8 * half - 2 * CCH;
                vbuf[((size_t)b * CCH + oc_a) * HW + n]      = (bf16_t)(acc0[t][r] + bias0[r]);
                vbuf[((size_t)b * CCH + oc_a + 16) * HW + n] = (bf16_t)(acc1[t][r] + bias1[r]);
            }
        }
    }
}

// ---------- GEMM2: scores = (q*scale)^T k, fused row softmax -> attn (bf16) ----------
// 512 threads = 16 waves. Block: (b, 32-row strip) x full 1024 cols.
// q strip (32x512) staged in LDS (padded stride -> conflict-free fragment reads).
#define QLD 520
__global__ void scores_softmax_kernel(const bf16_t* __restrict__ qkT,
                                      bf16_t* __restrict__ attn) {
    int b    = blockIdx.x;
    int n0   = blockIdx.y * 32;
    int wave = threadIdx.x >> 5;
    int lane = threadIdx.x & 31;
    int half = lane >> 4;

    __shared__ bf16_t qs[32 * QLD];
    __shared__ float  red[16][32];

    const bf16_t* qbase = qkT + ((size_t)(b * HW + n0)) * 1024;   // q cols 0..511
    const bf16_t* kbase = qkT + (size_t)b * HW * 1024 + CCH;      // k cols 512..1023

    // cooperative stage: thread i copies 32 elements of row i>>4, seg i&15
    {
        int row = threadIdx.x >> 4, seg = threadIdx.x & 15;
        const bf16_t* src = qbase + (size_t)row * 1024 + seg * 32;
        bf16_t* dst = qs + row * QLD + seg * 32;
        *(v8bf*)(dst)      = *(const v8bf*)(src);
        *(v8bf*)(dst + 8)  = *(const v8bf*)(src + 8);
        *(v8bf*)(dst + 16) = *(const v8bf*)(src + 16);
        *(v8bf*)(dst + 24) = *(const v8bf*)(src + 24);
    }
    __syncthreads();

    int mw  = wave * 64;
    int row = lane & 15;

    v8f acc0[4] = {};
    v8f acc1[4] = {};
#pragma unroll 1
    for (int k = 0; k < CCH; k += 32) {
        const bf16_t* p0 = qs + row * QLD + half * 8 + k;
        const bf16_t* p1 = p0 + 16 * QLD;
        v16bf a0 = make_a(*(const v8bf*)p0, *(const v8bf*)(p0 + 16));
        v16bf a1 = make_a(*(const v8bf*)p1, *(const v8bf*)(p1 + 16));
#pragma unroll
        for (int t = 0; t < 4; ++t) {
            v16bf bv = load_b_frag(kbase + (size_t)(mw + t * 16) * 1024 + k, 1024, lane);
            acc0[t] = wmma_bf16(a0, bv, acc0[t]);
            acc1[t] = wmma_bf16(a1, bv, acc1[t]);
        }
    }

    int rr = 8 * half;

    // ---- row max ----
    float m0[8], m1[8];
#pragma unroll
    for (int r = 0; r < 8; ++r) {
        float a = acc0[0][r], c = acc1[0][r];
#pragma unroll
        for (int t = 1; t < 4; ++t) { a = fmaxf(a, acc0[t][r]); c = fmaxf(c, acc1[t][r]); }
        m0[r] = half16_max(a);
        m1[r] = half16_max(c);
    }
    if ((lane & 15) == 0) {
#pragma unroll
        for (int r = 0; r < 8; ++r) {
            red[wave][r + rr]      = m0[r];
            red[wave][16 + r + rr] = m1[r];
        }
    }
    __syncthreads();
    float rmax0[8], rmax1[8];
#pragma unroll
    for (int r = 0; r < 8; ++r) {
        float a = red[0][r + rr], c = red[0][16 + r + rr];
#pragma unroll
        for (int w = 1; w < 16; ++w) {
            a = fmaxf(a, red[w][r + rr]);
            c = fmaxf(c, red[w][16 + r + rr]);
        }
        rmax0[r] = a; rmax1[r] = c;
    }
    __syncthreads();

    // ---- exp + row sum ----
    float s0[8], s1[8];
#pragma unroll
    for (int r = 0; r < 8; ++r) {
        float sa = 0.f, sc = 0.f;
#pragma unroll
        for (int t = 0; t < 4; ++t) {
            float e0 = __expf(acc0[t][r] - rmax0[r]);
            float e1 = __expf(acc1[t][r] - rmax1[r]);
            acc0[t][r] = e0; acc1[t][r] = e1;
            sa += e0; sc += e1;
        }
        s0[r] = half16_sum(sa);
        s1[r] = half16_sum(sc);
    }
    if ((lane & 15) == 0) {
#pragma unroll
        for (int r = 0; r < 8; ++r) {
            red[wave][r + rr]      = s0[r];
            red[wave][16 + r + rr] = s1[r];
        }
    }
    __syncthreads();
    float rinv0[8], rinv1[8];
#pragma unroll
    for (int r = 0; r < 8; ++r) {
        float sa = red[0][r + rr], sc = red[0][16 + r + rr];
#pragma unroll
        for (int w = 1; w < 16; ++w) {
            sa += red[w][r + rr];
            sc += red[w][16 + r + rr];
        }
        rinv0[r] = __frcp_rn(sa);
        rinv1[r] = __frcp_rn(sc);
    }

    bf16_t* out = attn + ((size_t)(b * HW + n0)) * 1024;
#pragma unroll
    for (int t = 0; t < 4; ++t) {
#pragma unroll
        for (int r = 0; r < 8; ++r) {
            int ro = r + rr;
            int c  = mw + t * 16 + (lane & 15);
            out[(size_t)ro * 1024 + c]        = (bf16_t)(acc0[t][r] * rinv0[r]);
            out[(size_t)(ro + 16) * 1024 + c] = (bf16_t)(acc1[t][r] * rinv1[r]);
        }
    }
}

// ---------- GEMM3: h[n][c] = sum_m attn[n][m] * v[c][m] ----------
// 256 threads = 8 waves. Block: (b, 32-row strip of n) x 512 c cols.
// attn strip (32x1024) staged in LDS.
#define ALD 1032
__global__ void attnv_kernel(const bf16_t* __restrict__ attn,
                             const bf16_t* __restrict__ vbuf,
                             bf16_t* __restrict__ hT) {
    int b    = blockIdx.x;
    int n0   = blockIdx.y * 32;
    int wave = threadIdx.x >> 5;
    int lane = threadIdx.x & 31;
    int half = lane >> 4;

    __shared__ bf16_t as[32 * ALD];

    const bf16_t* abase = attn + ((size_t)(b * HW + n0)) * 1024;
    const bf16_t* vb    = vbuf + (size_t)b * CCH * HW;

    // cooperative stage: thread i copies 128 elements of row i>>3, seg i&7
    {
        int row = threadIdx.x >> 3, seg = threadIdx.x & 7;
        const bf16_t* src = abase + (size_t)row * 1024 + seg * 128;
        bf16_t* dst = as + row * ALD + seg * 128;
#pragma unroll
        for (int j = 0; j < 16; ++j)
            *(v8bf*)(dst + j * 8) = *(const v8bf*)(src + j * 8);
    }
    __syncthreads();

    int cw  = wave * 64;
    int row = lane & 15;

    v8f acc0[4] = {};
    v8f acc1[4] = {};
#pragma unroll 1
    for (int k = 0; k < HW; k += 32) {
        const bf16_t* p0 = as + row * ALD + half * 8 + k;
        const bf16_t* p1 = p0 + 16 * ALD;
        v16bf a0 = make_a(*(const v8bf*)p0, *(const v8bf*)(p0 + 16));
        v16bf a1 = make_a(*(const v8bf*)p1, *(const v8bf*)(p1 + 16));
#pragma unroll
        for (int t = 0; t < 4; ++t) {
            v16bf bv = load_b_frag(vb + (size_t)(cw + t * 16) * HW + k, HW, lane);
            acc0[t] = wmma_bf16(a0, bv, acc0[t]);
            acc1[t] = wmma_bf16(a1, bv, acc1[t]);
        }
    }

    bf16_t* out = hT + ((size_t)(b * HW + n0)) * CCH;
#pragma unroll
    for (int t = 0; t < 4; ++t) {
#pragma unroll
        for (int r = 0; r < 8; ++r) {
            int ro = r + 8 * half;
            int c  = cw + t * 16 + (lane & 15);
            out[(size_t)ro * CCH + c]        = (bf16_t)acc0[t][r];
            out[(size_t)(ro + 16) * CCH + c] = (bf16_t)acc1[t][r];
        }
    }
}

// ---------- GEMM4: out = x + b_proj + w_proj @ h ----------
// 512 threads = 16 waves. Block: (b, 32-row strip of c') x full 1024 pixels.
// w_proj strip (32x512) staged in LDS.
__global__ void proj_residual_kernel(const bf16_t* __restrict__ wpb,
                                     const bf16_t* __restrict__ hT,
                                     const float*  __restrict__ x,
                                     const float*  __restrict__ b_proj,
                                     float* __restrict__ out) {
    int b    = blockIdx.x;
    int c0   = blockIdx.y * 32;
    int wave = threadIdx.x >> 5;
    int lane = threadIdx.x & 31;
    int half = lane >> 4;

    __shared__ bf16_t wsd[32 * QLD];

    const bf16_t* abase = wpb + (size_t)c0 * CCH;
    const bf16_t* hb    = hT + (size_t)b * HW * CCH;

    {
        int row = threadIdx.x >> 4, seg = threadIdx.x & 15;
        const bf16_t* src = abase + (size_t)row * CCH + seg * 32;
        bf16_t* dst = wsd + row * QLD + seg * 32;
        *(v8bf*)(dst)      = *(const v8bf*)(src);
        *(v8bf*)(dst + 8)  = *(const v8bf*)(src + 8);
        *(v8bf*)(dst + 16) = *(const v8bf*)(src + 16);
        *(v8bf*)(dst + 24) = *(const v8bf*)(src + 24);
    }
    __syncthreads();

    int nw  = wave * 64;
    int row = lane & 15;

    v8f acc0[4] = {};
    v8f acc1[4] = {};
#pragma unroll 1
    for (int k = 0; k < CCH; k += 32) {
        const bf16_t* p0 = wsd + row * QLD + half * 8 + k;
        const bf16_t* p1 = p0 + 16 * QLD;
        v16bf a0 = make_a(*(const v8bf*)p0, *(const v8bf*)(p0 + 16));
        v16bf a1 = make_a(*(const v8bf*)p1, *(const v8bf*)(p1 + 16));
#pragma unroll
        for (int t = 0; t < 4; ++t) {
            v16bf bv = load_b_frag(hb + (size_t)(nw + t * 16) * CCH + k, CCH, lane);
            acc0[t] = wmma_bf16(a0, bv, acc0[t]);
            acc1[t] = wmma_bf16(a1, bv, acc1[t]);
        }
    }

    float bias0[8], bias1[8];
#pragma unroll
    for (int r = 0; r < 8; ++r) {
        bias0[r] = b_proj[c0 + r + 8 * half];
        bias1[r] = b_proj[c0 + 16 + r + 8 * half];
    }

#pragma unroll
    for (int t = 0; t < 4; ++t) {
#pragma unroll
        for (int r = 0; r < 8; ++r) {
            int cp = c0 + r + 8 * half;
            int n  = nw + t * 16 + (lane & 15);
            size_t idx0 = ((size_t)b * CCH + cp) * HW + n;
            size_t idx1 = ((size_t)b * CCH + cp + 16) * HW + n;
            out[idx0] = x[idx0] + bias0[r] + acc0[t][r];
            out[idx1] = x[idx1] + bias1[r] + acc1[t][r];
        }
    }
}

// ---------- launch ----------

extern "C" void kernel_launch(void* const* d_in, const int* in_sizes, int n_in,
                              void* d_out, int out_size, void* d_ws, size_t ws_size,
                              hipStream_t stream) {
    const float* x      = (const float*)d_in[0];
    const float* w_qkv  = (const float*)d_in[1];
    const float* b_qkv  = (const float*)d_in[2];
    const float* w_proj = (const float*)d_in[3];
    const float* b_proj = (const float*)d_in[4];
    // d_in[5], d_in[6]: gn_gamma/gn_beta — dead in the reference graph.
    float* out = (float*)d_out;
    char*  ws  = (char*)d_ws;

    // workspace layout (bytes)
    const size_t o_xb   = 0;                                    // bf16 [B][PHW][C]  ~19 MB
    const size_t o_wq   = o_xb   + (size_t)NB * PHW * CCH * 2;  // bf16 [9][3C][C]  13.5 MB
    const size_t o_wp   = o_wq   + (size_t)9 * OC3 * CCH * 2;   // bf16 [C][C]       0.5 MB
    const size_t o_qkT  = o_wp   + (size_t)CCH * CCH * 2;       // bf16 [B][HW][2C]   32 MB
    const size_t o_v    = o_qkT  + (size_t)NB * HW * 1024 * 2;  // bf16 [B][C][HW]    16 MB
    const size_t o_attn = o_v    + (size_t)NB * CCH * HW * 2;   // bf16 [B][HW][HW]   32 MB
    const size_t o_hT   = o_attn + (size_t)NB * HW * HW * 2;    // bf16 [B][HW][C]    16 MB

    bf16_t* xb    = (bf16_t*)(ws + o_xb);
    bf16_t* wqkvb = (bf16_t*)(ws + o_wq);
    bf16_t* wpb   = (bf16_t*)(ws + o_wp);
    bf16_t* qkT   = (bf16_t*)(ws + o_qkT);
    bf16_t* vbuf  = (bf16_t*)(ws + o_v);
    bf16_t* attn  = (bf16_t*)(ws + o_attn);
    bf16_t* hT    = (bf16_t*)(ws + o_hT);

    const size_t n_x  = (size_t)NB * PHW * CCH;
    const size_t n_wq = (size_t)OC3 * CCH * 9;
    const size_t n_wp = (size_t)CCH * CCH;

    pack_x_kernel   <<<(unsigned)((n_x  + 255) / 256), 256, 0, stream>>>(x, xb);
    pack_wqkv_kernel<<<(unsigned)((n_wq + 255) / 256), 256, 0, stream>>>(w_qkv, wqkvb);
    pack_wp_kernel  <<<(unsigned)((n_wp + 255) / 256), 256, 0, stream>>>(w_proj, wpb);

    qkv_conv_kernel      <<<dim3(NB, OC3 / 32), 512, 0, stream>>>(xb, wqkvb, b_qkv, qkT, vbuf);
    scores_softmax_kernel<<<dim3(NB, HW  / 32), 512, 0, stream>>>(qkT, attn);
    attnv_kernel         <<<dim3(NB, HW  / 32), 256, 0, stream>>>(attn, vbuf, hT);
    proj_residual_kernel <<<dim3(NB, CCH / 32), 512, 0, stream>>>(wpb, hT, x, b_proj, out);
}
